// JitLSTMCell_59528246722716
// MI455X (gfx1250) — compile-verified, matrix operand order checked by reading, not run on hardware
//
#include <hip/hip_runtime.h>
#include <hip/hip_bf16.h>

typedef __bf16 bf16_t;
typedef __attribute__((ext_vector_type(16))) __bf16 v16bf;
typedef __attribute__((ext_vector_type(8)))  __bf16 v8bf;
typedef __attribute__((ext_vector_type(8)))  float  v8f;

#define B_  64
#define T_  512
#define I_  1024
#define H_  1024
#define K_  (I_ + H_)      // 2048 combined K
#define N4H (4 * H_)       // 4096

__device__ __forceinline__ float sigmoid_f(float x) {
    return 1.0f / (1.0f + __expf(-x));
}

__device__ __forceinline__ v8f splat8(float v) {
    v8f r;
#pragma unroll
    for (int i = 0; i < 8; ++i) r[i] = v;
    return r;
}

// ---------------------------------------------------------------------------
// Prep: build combined bf16 weight matrix Wc[4096][2048] = [W_ih | W_hh] (row-
// major over output-gate index n), and fused bias[n] = bias_ih[n]+bias_hh[n].
// ---------------------------------------------------------------------------
__global__ __launch_bounds__(256)
void lstm_prep_weights(const float* __restrict__ wih,   // [4096][1024]
                       const float* __restrict__ whh,   // [4096][1024]
                       const float* __restrict__ bih,   // [4096]
                       const float* __restrict__ bhh,   // [4096]
                       bf16_t* __restrict__ Wc,         // [4096][2048]
                       float* __restrict__ bias)        // [4096]
{
    size_t idx = (size_t)blockIdx.x * 256 + threadIdx.x;   // over 4096*2048
    int n = (int)(idx >> 11);
    int k = (int)(idx & (K_ - 1));
    float v = (k < I_) ? wih[(size_t)n * I_ + k] : whh[(size_t)n * H_ + (k - I_)];
    Wc[idx] = (bf16_t)v;
    if (k == 0) bias[n] = bih[n] + bhh[n];
}

// ---------------------------------------------------------------------------
// Init: stage A-operand for t=0 (bf16 x_0 | bf16 hx), c state = cx, and copy
// the reference's state outputs (the INITIAL hx, cx) to the tail of d_out.
// ---------------------------------------------------------------------------
__global__ __launch_bounds__(256)
void lstm_init(const float* __restrict__ x,    // [B][T][I]
               const float* __restrict__ hx,   // [B][H]
               const float* __restrict__ cx,   // [B][H]
               bf16_t* __restrict__ A_buf,     // [B][2048]
               float* __restrict__ c_ws,       // [B][H]
               float* __restrict__ out)        // hidden_seq | hx | cx
{
    int idx = blockIdx.x * 256 + threadIdx.x;  // < B*H = 65536
    int b = idx >> 10;
    int j = idx & (H_ - 1);
    A_buf[(size_t)b * K_ + j]      = (bf16_t)x[((size_t)b * T_) * I_ + j];  // x[b,0,j]
    A_buf[(size_t)b * K_ + I_ + j] = (bf16_t)hx[idx];
    c_ws[idx] = cx[idx];
    size_t seq = (size_t)B_ * T_ * H_;
    out[seq + idx]             = hx[idx];
    out[seq + B_ * H_ + idx]   = cx[idx];
}

// ---------------------------------------------------------------------------
// Per-step GEMM: gates[64][4096] = A_buf[64][2048](bf16) * Wc^T + bias.
// WMMA bf16 16x16x32, fp32 accumulate.  Workgroup = 4 waves (128 thr); wave w
// owns M-tile w (rows 16w..16w+15) and a 16x32 slab of N.  Grid = 4096/32.
// Fragment layout (per CDNA5 ISA 7.12.2, 16-bit A, wave32):
//   lane&15 = row (A) / col (B);  lane>>4 selects K-halves {koff..koff+7} and
//   {16+koff..16+koff+7} with koff = (lane>>4)*8  -> two contiguous 16B loads.
// ---------------------------------------------------------------------------
__global__ __launch_bounds__(128)
void lstm_gemm_step(const bf16_t* __restrict__ A_buf,  // [64][2048]
                    const bf16_t* __restrict__ Wc,     // [4096][2048]
                    const float*  __restrict__ bias,   // [4096]
                    float* __restrict__ gates)         // [64][4096]
{
    const int lane  = threadIdx.x & 31;
    const int wave  = threadIdx.x >> 5;      // 0..3 -> M tile
    const int m0    = wave * 16;
    const int n0    = blockIdx.x * 32;       // 2 N-tiles per wave
    const int r     = lane & 15;
    const int khalf = lane >> 4;
    const int koff  = khalf * 8;

    const bf16_t* Arow  = A_buf + (size_t)(m0 + r) * K_;
    const bf16_t* Brow0 = Wc + (size_t)(n0 + r) * K_;
    const bf16_t* Brow1 = Wc + (size_t)(n0 + 16 + r) * K_;

    v8f acc0 = splat8(bias[n0 + r]);
    v8f acc1 = splat8(bias[n0 + 16 + r]);

    for (int kc = 0; kc < K_; kc += 32) {
        // Weights sit in GL2 after step 0; locality=3 -> WGP-scope prefetch
        // ("pull into all cache levels"), staging the stream into WGP$ ahead
        // of the b128 loads.  (loc=0 lowered to SCOPE_SYS = GL2-only, useless.)
        __builtin_prefetch(Brow0 + kc + 128, 0, 3);
        __builtin_prefetch(Brow1 + kc + 128, 0, 3);

        v8bf alo = *(const v8bf*)(Arow + kc + koff);
        v8bf ahi = *(const v8bf*)(Arow + kc + 16 + koff);
        v8bf b0l = *(const v8bf*)(Brow0 + kc + koff);
        v8bf b0h = *(const v8bf*)(Brow0 + kc + 16 + koff);
        v8bf b1l = *(const v8bf*)(Brow1 + kc + koff);
        v8bf b1h = *(const v8bf*)(Brow1 + kc + 16 + koff);

        v16bf afrag, bfrag0, bfrag1;
#pragma unroll
        for (int e = 0; e < 8; ++e) {
            afrag[e]  = alo[e];  afrag[e + 8]  = ahi[e];
            bfrag0[e] = b0l[e];  bfrag0[e + 8] = b0h[e];
            bfrag1[e] = b1l[e];  bfrag1[e + 8] = b1h[e];
        }

        acc0 = __builtin_amdgcn_wmma_f32_16x16x32_bf16(
            false, afrag, false, bfrag0, (short)0, acc0, false, false);
        acc1 = __builtin_amdgcn_wmma_f32_16x16x32_bf16(
            false, afrag, false, bfrag1, (short)0, acc1, false, false);
    }

    // C/D layout: lane holds column N = n_tile + (lane&15); element e is row
    // M = m0 + khalf*8 + e.
#pragma unroll
    for (int e = 0; e < 8; ++e) {
        int m = m0 + khalf * 8 + e;
        gates[(size_t)m * N4H + n0 + r]      = acc0[e];
        gates[(size_t)m * N4H + n0 + 16 + r] = acc1[e];
    }
}

// ---------------------------------------------------------------------------
// Per-step pointwise: peephole gate math, c/h update, write hy to hidden_seq,
// and stage next step's A-operand (bf16 h and bf16 x[t+1]).
// ---------------------------------------------------------------------------
__global__ __launch_bounds__(256)
void lstm_pointwise(const float* __restrict__ gates,  // [64][4096]
                    float* __restrict__ c_ws,         // [64][1024]
                    bf16_t* __restrict__ A_buf,       // [64][2048]
                    const float* __restrict__ x,      // [B][T][I]
                    const float* __restrict__ wci,
                    const float* __restrict__ wcf,
                    const float* __restrict__ wco,
                    float* __restrict__ out_seq,      // [B][T][H]
                    int t)
{
    int idx = blockIdx.x * 256 + threadIdx.x;  // < B*H
    int b = idx >> 10;
    int j = idx & (H_ - 1);

    const float* g = gates + (size_t)b * N4H;
    float c  = c_ws[idx];
    float gi = g[j];
    float gf = g[j + H_];
    float gg = g[j + 2 * H_];
    float go = g[j + 3 * H_];

    float i_g = sigmoid_f(gi + wci[j] * c);
    float f_g = sigmoid_f(gf + wcf[j] * c);
    float g_g = tanhf(gg);
    float cy  = f_g * c + i_g * g_g;
    float o_g = sigmoid_f(go + wco[j] * cy);
    float hy  = o_g * tanhf(cy);

    c_ws[idx] = cy;
    A_buf[(size_t)b * K_ + I_ + j] = (bf16_t)hy;
    if (t + 1 < T_)
        A_buf[(size_t)b * K_ + j] = (bf16_t)x[((size_t)b * T_ + (t + 1)) * I_ + j];
    out_seq[((size_t)b * T_ + t) * H_ + j] = hy;
}

// ---------------------------------------------------------------------------
extern "C" void kernel_launch(void* const* d_in, const int* in_sizes, int n_in,
                              void* d_out, int out_size, void* d_ws, size_t ws_size,
                              hipStream_t stream) {
    const float* x   = (const float*)d_in[0];
    const float* hx  = (const float*)d_in[1];
    const float* cx  = (const float*)d_in[2];
    const float* wih = (const float*)d_in[3];
    const float* whh = (const float*)d_in[4];
    const float* bih = (const float*)d_in[5];
    const float* bhh = (const float*)d_in[6];
    const float* wci = (const float*)d_in[7];
    const float* wcf = (const float*)d_in[8];
    const float* wco = (const float*)d_in[9];
    float* out = (float*)d_out;

    // Workspace layout (bytes, 256-aligned)
    char* ws = (char*)d_ws;
    bf16_t* Wc    = (bf16_t*)(ws);                        // 4096*2048*2 = 16 MB
    float*  bias  = (float*)(ws + 16777216);              // 16 KB
    bf16_t* A_buf = (bf16_t*)(ws + 16793600);             // 64*2048*2 = 256 KB
    float*  c_ws  = (float*)(ws + 17055744);              // 64*1024*4 = 256 KB
    float*  gates = (float*)(ws + 17317888);              // 64*4096*4 = 1 MB

    // One-time prep (weights to bf16, fused bias)
    lstm_prep_weights<<<(N4H * K_) / 256, 256, 0, stream>>>(wih, whh, bih, bhh, Wc, bias);
    // Init A staging (x_0 | hx), c state, and state outputs (initial hx, cx)
    lstm_init<<<(B_ * H_) / 256, 256, 0, stream>>>(x, hx, cx, A_buf, c_ws, out);

    for (int t = 0; t < T_; ++t) {
        lstm_gemm_step<<<N4H / 32, 128, 0, stream>>>(A_buf, Wc, bias, gates);
        lstm_pointwise<<<(B_ * H_) / 256, 256, 0, stream>>>(
            gates, c_ws, A_buf, x, wci, wcf, wco, out, t);
    }
}
